// DruseScorePKiForExport_39359080300922
// MI455X (gfx1250) — matrix operand order repaired and checked by reading, not compiled, not run on hardware
//
#include <hip/hip_runtime.h>
#include <hip/hip_bf16.h>

#define L_   256
#define P_   1024
#define D_   256
#define NH_  4
#define RBF_ 50
#define NL_  2
#define HD_  64
#define LP_  (L_*P_)
#define EPS_ 1e-5f

typedef __attribute__((ext_vector_type(16))) _Float16 v16h;
typedef __attribute__((ext_vector_type(8)))  _Float16 v8h;
typedef __attribute__((ext_vector_type(8)))  float    v8f;
typedef __attribute__((ext_vector_type(4)))  int      v4i;

#define AS1 __attribute__((address_space(1)))
#define AS3 __attribute__((address_space(3)))

// gfx1250 async global->LDS path (ASYNCcnt). Guarded: falls back to plain LDS staging
// if the toolchain doesn't declare the builtins (host pass always takes fallback).
// Probe round 2 confirmed the signature: (v4i AS1*, v4i AS3*, imm int, imm int).
#if __has_builtin(__builtin_amdgcn_global_load_async_to_lds_b128) && __has_builtin(__builtin_amdgcn_s_wait_asynccnt)
#define USE_ASYNC_LDS 1
#else
#define USE_ASYNC_LDS 0
#endif

#if USE_ASYNC_LDS
// Copy 16 bytes global -> LDS asynchronously (per-lane addresses).
__device__ __forceinline__ void async_cp16(const void* gsrc, void* lds_dst) {
  __builtin_amdgcn_global_load_async_to_lds_b128((AS1 v4i*)gsrc, (AS3 v4i*)lds_dst, 0, 0);
}
#endif

__device__ __forceinline__ float siluf(float x) { return x / (1.0f + __expf(-x)); }
__device__ __forceinline__ float sigf (float x) { return 1.0f / (1.0f + __expf(-x)); }

union HU { v16h v; v8h h[2]; };

// A fragment: 16x32 f16, row-major source, per ISA 7.12.2:
// lanes 0-15 hold K={0..7,16..23}, lanes 16-31 hold K={8..15,24..31}; row = lane%16.
__device__ __forceinline__ v16h load_a_frag(const _Float16* __restrict__ A, int lda,
                                            int row, int kBase, int lane) {
  int hv = lane >> 4;
  const _Float16* p = A + (size_t)row * lda + kBase + hv * 8;
  HU u;
  u.h[0] = *(const v8h*)(p);
  u.h[1] = *(const v8h*)(p + 16);
  return u.v;
}

// B fragment from Bt stored [N,K] row-major: lane n%16 = column,
// lanes 0-15 hold K={0..7,16..23}, lanes 16-31 hold K={8..15,24..31} (matches A).
__device__ __forceinline__ v16h load_b_frag(const _Float16* __restrict__ Bt, int ldbt,
                                            int col, int kBase, int lane) {
  int hv = lane >> 4;
  const _Float16* p = Bt + (size_t)col * ldbt + kBase + hv * 8;
  HU u;
  u.h[0] = *(const v8h*)(p);
  u.h[1] = *(const v8h*)(p + 16);
  return u.v;
}

// Generic f16 WMMA GEMM: C[M,N] = A[M,K] @ Bt[N,K]^T (+bias).
// Block = 256 threads = 8 waves; tile 128x64; wave w -> rows m0+16w, 4 col-subtiles.
// OUTM: 0=f32, 1=f16, 2=f16 transposed (ldc = transposed leading dim), 3=f32+f16
// ACT : 0=none, 2=scale by 0.125 (Q/sqrt(HD))
template <int OUTM, int ACT, bool BIAS>
__global__ __launch_bounds__(256) void gemm16(
    const _Float16* __restrict__ A, int lda,
    const _Float16* __restrict__ Bt, int ldbt,
    const float* __restrict__ bias,
    float* __restrict__ Cf, _Float16* __restrict__ Ch, int ldc, int K)
{
  const int lane = threadIdx.x & 31;
  const int w    = threadIdx.x >> 5;
  const int m0   = blockIdx.x * 128 + w * 16;
  const int n0   = blockIdx.y * 64;
  const int r    = lane & 15;
  const v8f zero = {0.f,0.f,0.f,0.f,0.f,0.f,0.f,0.f};
  v8f acc[4] = { zero, zero, zero, zero };
  const int arow = m0 + r;
  for (int kk = 0; kk < K; kk += 32) {
    v16h a = load_a_frag(A, lda, arow, kk, lane);
    if (kk + 64 <= K) __builtin_prefetch(A + (size_t)arow * lda + kk + 32, 0, 1);
#pragma unroll
    for (int ct = 0; ct < 4; ++ct) {
      v16h b = load_b_frag(Bt, ldbt, n0 + ct * 16 + r, kk, lane);
      acc[ct] = __builtin_amdgcn_wmma_f32_16x16x32_f16(false, a, false, b,
                                                       (short)0, acc[ct], false, false);
    }
  }
  const int hv = lane >> 4;
#pragma unroll
  for (int ct = 0; ct < 4; ++ct) {
    const int n = n0 + ct * 16 + r;
    float bj = BIAS ? bias[n] : 0.0f;
#pragma unroll
    for (int v = 0; v < 8; ++v) {
      const int m = m0 + v + hv * 8;
      float x = acc[ct][v] + bj;
      if (ACT == 2) x *= 0.125f;
      if (OUTM == 0)      Cf[(size_t)m * ldc + n] = x;
      else if (OUTM == 1) Ch[(size_t)m * ldc + n] = (_Float16)x;
      else if (OUTM == 2) Ch[(size_t)n * ldc + m] = (_Float16)x;
      else { Cf[(size_t)m * ldc + n] = x; Ch[(size_t)m * ldc + n] = (_Float16)x; }
    }
  }
}

// Transpose-convert f32 [K,N] (row stride N) -> f16 [N,Kpad], zero-padded K..Kpad.
__global__ __launch_bounds__(256) void convertT_k(const float* __restrict__ src,
                                                  _Float16* __restrict__ dst,
                                                  int K, int N, int Kpad)
{
  int idx = blockIdx.x * 256 + threadIdx.x;
  if (idx >= N * Kpad) return;
  int n = idx / Kpad, k = idx - n * Kpad;
  dst[idx] = (k < K) ? (_Float16)src[(size_t)k * N + n] : (_Float16)0.0f;
}

// Encoder layer 1: enc16[row] = f16(silu(x[row,0:20] @ W1 + b1)), rows 0..1023 protein, 1024..1279 ligand.
__global__ __launch_bounds__(256) void encoder1_k(
    const float* __restrict__ pf, const float* __restrict__ lf,
    const float* __restrict__ pw1, const float* __restrict__ pb1,
    const float* __restrict__ lw1, const float* __restrict__ lb1,
    _Float16* __restrict__ enc16)
{
  __shared__ float xs[20];
  int row = blockIdx.x;
  const float *x, *w1, *b1;
  if (row < P_) { x = pf + (size_t)row * 20; w1 = pw1; b1 = pb1; }
  else          { x = lf + (size_t)(row - P_) * 20; w1 = lw1; b1 = lb1; }
  int t = threadIdx.x;
  if (t < 20) xs[t] = x[t];
  __syncthreads();
  float acc = b1[t];
  for (int k = 0; k < 20; ++k) acc += xs[k] * w1[k * D_ + t];
  enc16[(size_t)row * D_ + t] = (_Float16)siluf(acc);
}

// One pass over pair_rbf (52 MB): produce per-layer per-head attention biases (rbf@rw)
// and f16 rbf padded K=50 -> 64 for the WMMA pair GEMM. LDS fill uses async-to-LDS.
__global__ __launch_bounds__(256) void rbfprep_k(const float* __restrict__ rbf,
                                                 const float* __restrict__ rw,
                                                 float* __restrict__ rbias,
                                                 _Float16* __restrict__ rbf16)
{
  __shared__ __align__(16) float s[256 * RBF_];
  int t = threadIdx.x;
  size_t base = (size_t)blockIdx.x * 256;
#if USE_ASYNC_LDS
  // 256*50 floats = 3200 x 16B chunks, contiguous in global memory.
  for (int i = t; i < (256 * RBF_) / 4; i += 256)
    async_cp16(rbf + base * RBF_ + (size_t)i * 4, &s[i * 4]);
  __builtin_amdgcn_s_wait_asynccnt(0);
#else
  for (int i = t; i < 256 * RBF_; i += 256) s[i] = rbf[base * RBF_ + i];
#endif
  __syncthreads();
  float acc[8];
#pragma unroll
  for (int j = 0; j < 8; ++j) acc[j] = 0.f;
  const float* sr = s + t * RBF_;
  for (int rr = 0; rr < RBF_; ++rr) {
    float xv = sr[rr];
#pragma unroll
    for (int j = 0; j < 8; ++j) {
      int i = j >> 2, h = j & 3;
      acc[j] += xv * rw[(i * RBF_ + rr) * NH_ + h];
    }
  }
  size_t pair = base + t;
#pragma unroll
  for (int j = 0; j < 8; ++j) rbias[pair * 8 + j] = acc[j];
  for (int rr = 0; rr < 64; ++rr)
    rbf16[pair * 64 + rr] = (rr < RBF_) ? (_Float16)sr[rr] : (_Float16)0.0f;
}

// Softmax over P per (head, l) row: adds rbf bias + rb, writes f16 attn for WMMA and
// overwrites the f32 scores row in place with attn (for exact last-layer head mean).
__global__ __launch_bounds__(256) void softmax_k(float* __restrict__ scores,
                                                 const float* __restrict__ rbias,
                                                 const float* __restrict__ rb,
                                                 _Float16* __restrict__ attn16, int layer)
{
  __shared__ float red[256];
  int h = blockIdx.x >> 8;
  int l = blockIdx.x & 255;
  int t = threadIdx.x;
  float* srow = scores + ((size_t)h * L_ + l) * P_;
  float rbh = rb[layer * NH_ + h];
  float vals[4], mx = -3.0e38f;
#pragma unroll
  for (int i = 0; i < 4; ++i) {
    int p = t + i * 256;
    float sc = srow[p] + rbias[((size_t)l * P_ + p) * 8 + layer * 4 + h] + rbh;
    vals[i] = sc; mx = fmaxf(mx, sc);
  }
  red[t] = mx; __syncthreads();
  for (int s2 = 128; s2 > 0; s2 >>= 1) { if (t < s2) red[t] = fmaxf(red[t], red[t + s2]); __syncthreads(); }
  mx = red[0]; __syncthreads();
  float sum = 0.f;
#pragma unroll
  for (int i = 0; i < 4; ++i) { vals[i] = __expf(vals[i] - mx); sum += vals[i]; }
  red[t] = sum; __syncthreads();
  for (int s2 = 128; s2 > 0; s2 >>= 1) { if (t < s2) red[t] += red[t + s2]; __syncthreads(); }
  float inv = 1.0f / red[0];
  _Float16* arow = attn16 + ((size_t)h * L_ + l) * P_;
#pragma unroll
  for (int i = 0; i < 4; ++i) { int p = t + i * 256; float a = vals[i] * inv; srow[p] = a; arow[p] = (_Float16)a; }
}

// lig_h = LN(lig_h + oproj) * g + b ; also refresh f16 copy.
__global__ __launch_bounds__(256) void ln_k(float* __restrict__ lig, _Float16* __restrict__ lig16,
                                            const float* __restrict__ oproj,
                                            const float* __restrict__ g, const float* __restrict__ b)
{
  __shared__ float red[256];
  int l = blockIdx.x, t = threadIdx.x;
  float x = lig[l * D_ + t] + oproj[l * D_ + t];
  red[t] = x; __syncthreads();
  for (int s2 = 128; s2 > 0; s2 >>= 1) { if (t < s2) red[t] += red[t + s2]; __syncthreads(); }
  float mu = red[0] * (1.0f / D_); __syncthreads();
  float dd = x - mu;
  red[t] = dd * dd; __syncthreads();
  for (int s2 = 128; s2 > 0; s2 >>= 1) { if (t < s2) red[t] += red[t + s2]; __syncthreads(); }
  float var = red[0] * (1.0f / D_);
  float y = dd * rsqrtf(var + EPS_) * g[t] + b[t];
  lig[l * D_ + t] = y;
  lig16[l * D_ + t] = (_Float16)y;
}

__global__ __launch_bounds__(256) void attnmean_k(const float* __restrict__ attn,
                                                  float* __restrict__ outp)
{
  size_t pair = (size_t)blockIdx.x * 256 + threadIdx.x;
  float s = attn[pair] + attn[(size_t)LP_ + pair] + attn[2 * (size_t)LP_ + pair] + attn[3 * (size_t)LP_ + pair];
  outp[pair] = 0.25f * s;
}

// Gate logits: logits[l] = tanh(lig[l]@pg_w1 + pg_b1) @ pg_w2 + pg_b2
__global__ __launch_bounds__(128) void pool1_k(const float* __restrict__ lig, const float* __restrict__ w1,
                                               const float* __restrict__ b1, const float* __restrict__ w2,
                                               const float* __restrict__ b2, float* __restrict__ logits)
{
  __shared__ float red[128];
  int l = blockIdx.x, j = threadIdx.x;
  float acc = b1[j];
  for (int d = 0; d < D_; ++d) acc += lig[l * D_ + d] * w1[d * 128 + j];
  red[j] = tanhf(acc) * w2[j]; __syncthreads();
  for (int s2 = 64; s2 > 0; s2 >>= 1) { if (j < s2) red[j] += red[j + s2]; __syncthreads(); }
  if (j == 0) logits[l] = red[0] + b2[0];
}

// Gate softmax over L, gated pooling, pkd & conf heads, first 3 scalar outputs.
__global__ __launch_bounds__(256) void pool2_k(const float* __restrict__ lig, const float* __restrict__ logits,
    const float* __restrict__ afw1, const float* __restrict__ afb1,
    const float* __restrict__ afw2, const float* __restrict__ afb2,
    const float* __restrict__ cfw1, const float* __restrict__ cfb1,
    const float* __restrict__ cfw2, const float* __restrict__ cfb2,
    float* __restrict__ outp)
{
  __shared__ float red[256];
  __shared__ float gate[256];
  __shared__ float cvec[256];
  int t = threadIdx.x;
  float lg = logits[t];
  red[t] = lg; __syncthreads();
  for (int s2 = 128; s2 > 0; s2 >>= 1) { if (t < s2) red[t] = fmaxf(red[t], red[t + s2]); __syncthreads(); }
  float mx = red[0]; __syncthreads();
  float e = __expf(lg - mx);
  red[t] = e; __syncthreads();
  for (int s2 = 128; s2 > 0; s2 >>= 1) { if (t < s2) red[t] += red[t + s2]; __syncthreads(); }
  gate[t] = e / red[0]; __syncthreads();
  float c = 0.f;
  for (int l = 0; l < L_; ++l) c += gate[l] * lig[l * D_ + t];
  cvec[t] = c; __syncthreads();
  float acc = afb1[t];
  for (int d = 0; d < D_; ++d) acc += cvec[d] * afw1[d * D_ + t];
  red[t] = siluf(acc) * afw2[t]; __syncthreads();
  for (int s2 = 128; s2 > 0; s2 >>= 1) { if (t < s2) red[t] += red[t + s2]; __syncthreads(); }
  float pkd = red[0] + afb2[0]; __syncthreads();
  float p2 = 0.f;
  if (t < 128) {
    float a2 = cfb1[t];
    for (int d = 0; d < D_; ++d) a2 += cvec[d] * cfw1[d * 128 + t];
    p2 = siluf(a2) * cfw2[t];
  }
  red[t] = p2; __syncthreads();
  for (int s2 = 128; s2 > 0; s2 >>= 1) { if (t < s2) red[t] += red[t + s2]; __syncthreads(); }
  if (t == 0) {
    float conf = sigf(red[0] + cfb2[0]);
    outp[0] = pkd * conf;
    outp[1] = pkd;
    outp[2] = conf;
  }
}

// Fused pair kernel: WMMA rbf16 @ iw1_rbf^T, add A_lig[l]+B_prot[p]+ib1, silu,
// second layer (N=5) via per-lane partials + shfl reductions, sigmoid, store imap.
// Block = 32 pairs (all same l, P%32==0) x full 256 hidden cols; 8 waves (2 row x 4 col).
// LDS staging of the B_prot tile / row vectors / iw2 uses the async-to-LDS path.
__global__ __launch_bounds__(256) void imap_k(const _Float16* __restrict__ rbf16,
                                              const _Float16* __restrict__ wrbf,
                                              const float* __restrict__ Alig,
                                              const float* __restrict__ Bprot,
                                              const float* __restrict__ ib1,
                                              const float* __restrict__ iw2,
                                              const float* __restrict__ ib2,
                                              float* __restrict__ outp)
{
  __shared__ __align__(16) float sB[32 * D_];
  __shared__ __align__(16) float sA[D_];
  __shared__ __align__(16) float sb1[D_];
  __shared__ __align__(16) float sw2[D_ * 5];
  __shared__ float sred[4][32][5];
  int t = threadIdx.x;
  size_t r0 = (size_t)blockIdx.x * 32;
  int l  = (int)(r0 >> 10);
  int pb = (int)(r0 & 1023);
#if USE_ASYNC_LDS
  // B_prot tile: 32 full rows -> contiguous 32*256 floats = 2048 x 16B chunks.
  for (int i = t; i < (32 * D_) / 4; i += 256)
    async_cp16(Bprot + (size_t)pb * D_ + (size_t)i * 4, &sB[i * 4]);
  // iw2: 1280 floats = 320 chunks.
  for (int i = t; i < (D_ * 5) / 4; i += 256)
    async_cp16(iw2 + (size_t)i * 4, &sw2[i * 4]);
  if (t < D_ / 4) {
    async_cp16(Alig + (size_t)l * D_ + (size_t)t * 4, &sA[t * 4]);
    async_cp16(ib1 + (size_t)t * 4, &sb1[t * 4]);
  }
  __builtin_amdgcn_s_wait_asynccnt(0);
#else
  for (int i = t; i < 32 * D_; i += 256) sB[i] = Bprot[(size_t)(pb + (i >> 8)) * D_ + (i & 255)];
  if (t < D_) { sA[t] = Alig[l * D_ + t]; sb1[t] = ib1[t]; }
  for (int i = t; i < D_ * 5; i += 256) sw2[i] = iw2[i];
#endif
  __syncthreads();
  int lane = t & 31, w = t >> 5;
  int rowHalf = w >> 2;
  int nb = (w & 3) * 64;
  int mLoc = rowHalf * 16;
  int r = lane & 15, hv = lane >> 4;
  const v8f zero = {0.f,0.f,0.f,0.f,0.f,0.f,0.f,0.f};
  v8f acc[4] = { zero, zero, zero, zero };
  int arow = (int)r0 + mLoc + r;
  for (int kk = 0; kk < 64; kk += 32) {
    v16h a = load_a_frag(rbf16, 64, arow, kk, lane);
#pragma unroll
    for (int ct = 0; ct < 4; ++ct) {
      v16h b = load_b_frag(wrbf, 64, nb + ct * 16 + r, kk, lane);
      acc[ct] = __builtin_amdgcn_wmma_f32_16x16x32_f16(false, a, false, b,
                                                       (short)0, acc[ct], false, false);
    }
  }
  float part[8][5];
#pragma unroll
  for (int v = 0; v < 8; ++v)
#pragma unroll
    for (int c = 0; c < 5; ++c) part[v][c] = 0.f;
#pragma unroll
  for (int ct = 0; ct < 4; ++ct) {
    int j = nb + ct * 16 + r;
    float aj = sA[j] + sb1[j];
    float w2c[5];
#pragma unroll
    for (int c = 0; c < 5; ++c) w2c[c] = sw2[j * 5 + c];
#pragma unroll
    for (int v = 0; v < 8; ++v) {
      int lrow = mLoc + v + hv * 8;
      float x = siluf(acc[ct][v] + aj + sB[lrow * D_ + j]);
#pragma unroll
      for (int c = 0; c < 5; ++c) part[v][c] += x * w2c[c];
    }
  }
#pragma unroll
  for (int v = 0; v < 8; ++v)
#pragma unroll
    for (int c = 0; c < 5; ++c) {
      float s = part[v][c];
      for (int m = 1; m < 16; m <<= 1) s += __shfl_xor(s, m, 32);  // reduces within 16-lane halves
      part[v][c] = s;
    }
  if (r == 0) {
#pragma unroll
    for (int v = 0; v < 8; ++v) {
      int lrow = mLoc + v + hv * 8;
#pragma unroll
      for (int c = 0; c < 5; ++c) sred[w & 3][lrow][c] = part[v][c];
    }
  }
  __syncthreads();
  if (t < 160) {
    int row = t / 5, c = t - row * 5;
    float s = sred[0][row][c] + sred[1][row][c] + sred[2][row][c] + sred[3][row][c] + ib2[c];
    outp[(r0 + row) * 5 + c] = sigf(s);
  }
}

extern "C" void kernel_launch(void* const* d_in, const int* in_sizes, int n_in,
                              void* d_out, int out_size, void* d_ws, size_t ws_size,
                              hipStream_t stream)
{
  (void)in_sizes; (void)n_in; (void)out_size; (void)ws_size;
  const float* pf    = (const float*)d_in[0];
  const float* lf    = (const float*)d_in[1];
  const float* rbf   = (const float*)d_in[4];
  const float* pe_w1 = (const float*)d_in[5];
  const float* pe_b1 = (const float*)d_in[6];
  const float* pe_w2 = (const float*)d_in[7];
  const float* pe_b2 = (const float*)d_in[8];
  const float* le_w1 = (const float*)d_in[9];
  const float* le_b1 = (const float*)d_in[10];
  const float* le_w2 = (const float*)d_in[11];
  const float* le_b2 = (const float*)d_in[12];
  const float* qw = (const float*)d_in[13];
  const float* qb = (const float*)d_in[14];
  const float* kw = (const float*)d_in[15];
  const float* kb = (const float*)d_in[16];
  const float* vw = (const float*)d_in[17];
  const float* vb = (const float*)d_in[18];
  const float* rw = (const float*)d_in[19];
  const float* rb = (const float*)d_in[20];
  const float* ow = (const float*)d_in[21];
  const float* ob = (const float*)d_in[22];
  const float* ln_g = (const float*)d_in[23];
  const float* ln_b = (const float*)d_in[24];
  const float* pg_w1 = (const float*)d_in[25];
  const float* pg_b1 = (const float*)d_in[26];
  const float* pg_w2 = (const float*)d_in[27];
  const float* pg_b2 = (const float*)d_in[28];
  const float* af_w1 = (const float*)d_in[29];
  const float* af_b1 = (const float*)d_in[30];
  const float* af_w2 = (const float*)d_in[31];
  const float* af_b2 = (const float*)d_in[32];
  const float* cf_w1 = (const float*)d_in[33];
  const float* cf_b1 = (const float*)d_in[34];
  const float* cf_w2 = (const float*)d_in[35];
  const float* cf_b2 = (const float*)d_in[36];
  const float* iw1 = (const float*)d_in[37];
  const float* ib1 = (const float*)d_in[38];
  const float* iw2 = (const float*)d_in[39];
  const float* ib2 = (const float*)d_in[40];
  float* outp = (float*)d_out;

  char* wsbase = (char*)d_ws;
  size_t off = 0;
  auto alloc = [&](size_t bytes) -> char* {
    char* p = wsbase + off;
    off += (bytes + 255) & ~(size_t)255;
    return p;
  };
  _Float16* enc16  = (_Float16*)alloc((size_t)(P_ + L_) * D_ * 2);
  _Float16* prot16 = (_Float16*)alloc((size_t)P_ * D_ * 2);
  float*    ligH   = (float*)   alloc((size_t)L_ * D_ * 4);
  _Float16* lig16  = (_Float16*)alloc((size_t)L_ * D_ * 2);
  _Float16* wT     = (_Float16*)alloc((size_t)8 * D_ * D_ * 2);  // [layer][q,k,v,o] transposed f16
  _Float16* pe2T   = (_Float16*)alloc((size_t)D_ * D_ * 2);
  _Float16* le2T   = (_Float16*)alloc((size_t)D_ * D_ * 2);
  _Float16* w1lT   = (_Float16*)alloc((size_t)D_ * D_ * 2);
  _Float16* w1pT   = (_Float16*)alloc((size_t)D_ * D_ * 2);
  _Float16* w1rT   = (_Float16*)alloc((size_t)D_ * 64 * 2);
  _Float16* Q16    = (_Float16*)alloc((size_t)L_ * D_ * 2);
  _Float16* K16    = (_Float16*)alloc((size_t)P_ * D_ * 2);
  _Float16* Vt16   = (_Float16*)alloc((size_t)D_ * P_ * 2);     // transposed: [d][p]
  float*    scores = (float*)   alloc((size_t)NH_ * LP_ * 4);   // [h][l][p]
  _Float16* attn16 = (_Float16*)alloc((size_t)NH_ * LP_ * 2);
  _Float16* att16  = (_Float16*)alloc((size_t)L_ * D_ * 2);
  float*    oproj  = (float*)   alloc((size_t)L_ * D_ * 4);
  float*    rbias  = (float*)   alloc((size_t)LP_ * 8 * 4);
  _Float16* rbf16  = (_Float16*)alloc((size_t)LP_ * 64 * 2);
  float*    Alig   = (float*)   alloc((size_t)L_ * D_ * 4);
  float*    Bprot  = (float*)   alloc((size_t)P_ * D_ * 4);
  float*    logits = (float*)   alloc((size_t)L_ * 4);

  auto cvt = [&](const float* src, _Float16* dst, int K, int N, int Kpad) {
    int total = N * Kpad;
    convertT_k<<<(total + 255) / 256, 256, 0, stream>>>(src, dst, K, N, Kpad);
  };
  cvt(pe_w2, pe2T, D_, D_, D_);
  cvt(le_w2, le2T, D_, D_, D_);
  for (int i = 0; i < NL_; ++i) {
    cvt(qw + (size_t)i * D_ * D_, wT + ((size_t)i * 4 + 0) * D_ * D_, D_, D_, D_);
    cvt(kw + (size_t)i * D_ * D_, wT + ((size_t)i * 4 + 1) * D_ * D_, D_, D_, D_);
    cvt(vw + (size_t)i * D_ * D_, wT + ((size_t)i * 4 + 2) * D_ * D_, D_, D_, D_);
    cvt(ow + (size_t)i * D_ * D_, wT + ((size_t)i * 4 + 3) * D_ * D_, D_, D_, D_);
  }
  cvt(iw1,             w1lT, D_,   D_, D_);
  cvt(iw1 + 256 * D_,  w1pT, D_,   D_, D_);
  cvt(iw1 + 512 * D_,  w1rT, RBF_, D_, 64);

  // Encoders
  encoder1_k<<<P_ + L_, 256, 0, stream>>>(pf, lf, pe_w1, pe_b1, le_w1, le_b1, enc16);
  gemm16<1,0,true><<<dim3(P_/128, D_/64), 256, 0, stream>>>(enc16, D_, pe2T, D_, pe_b2, nullptr, prot16, D_, D_);
  gemm16<3,0,true><<<dim3(L_/128, D_/64), 256, 0, stream>>>(enc16 + (size_t)P_ * D_, D_, le2T, D_, le_b2, ligH, lig16, D_, D_);

  // Single pass over pair_rbf: attention biases for both layers + f16 padded copy
  rbfprep_k<<<LP_/256, 256, 0, stream>>>(rbf, rw, rbias, rbf16);

  for (int i = 0; i < NL_; ++i) {
    const _Float16* qT = wT + ((size_t)i * 4 + 0) * D_ * D_;
    const _Float16* kT = wT + ((size_t)i * 4 + 1) * D_ * D_;
    const _Float16* vT = wT + ((size_t)i * 4 + 2) * D_ * D_;
    const _Float16* oT = wT + ((size_t)i * 4 + 3) * D_ * D_;
    gemm16<1,2,true><<<dim3(L_/128, D_/64), 256, 0, stream>>>(lig16,  D_, qT, D_, qb + i * D_, nullptr, Q16, D_, D_);
    gemm16<1,0,true><<<dim3(P_/128, D_/64), 256, 0, stream>>>(prot16, D_, kT, D_, kb + i * D_, nullptr, K16, D_, D_);
    gemm16<2,0,true><<<dim3(P_/128, D_/64), 256, 0, stream>>>(prot16, D_, vT, D_, vb + i * D_, nullptr, Vt16, P_, D_);
    for (int h = 0; h < NH_; ++h)
      gemm16<0,0,false><<<dim3(L_/128, P_/64), 256, 0, stream>>>(Q16 + h * HD_, D_, K16 + h * HD_, D_,
                                                                 nullptr, scores + (size_t)h * LP_, nullptr, P_, HD_);
    softmax_k<<<NH_ * L_, 256, 0, stream>>>(scores, rbias, rb, attn16, i);
    for (int h = 0; h < NH_; ++h)
      gemm16<1,0,false><<<dim3(L_/128, HD_/64), 256, 0, stream>>>(attn16 + (size_t)h * LP_, P_,
                                                                  Vt16 + (size_t)h * HD_ * P_, P_,
                                                                  nullptr, nullptr, att16 + h * HD_, D_, P_);
    gemm16<0,0,true><<<dim3(L_/128, D_/64), 256, 0, stream>>>(att16, D_, oT, D_, ob + i * D_, oproj, nullptr, D_, D_);
    ln_k<<<L_, 256, 0, stream>>>(ligH, lig16, oproj, ln_g + i * D_, ln_b + i * D_);
  }

  // attn_out = mean over heads of last-layer attn (exact f32, from in-place buffer)
  attnmean_k<<<LP_/256, 256, 0, stream>>>(scores, outp + 3 + (size_t)5 * LP_);

  // Gated pooling + scalar heads
  pool1_k<<<L_, 128, 0, stream>>>(ligH, pg_w1, pg_b1, pg_w2, pg_b2, logits);
  pool2_k<<<1, 256, 0, stream>>>(ligH, logits, af_w1, af_b1, af_w2, af_b2,
                                 cf_w1, cf_b1, cf_w2, cf_b2, outp);

  // Interaction map: precompute row/col terms, then fused per-pair WMMA kernel
  gemm16<0,0,false><<<dim3(L_/128, D_/64), 256, 0, stream>>>(lig16,  D_, w1lT, D_, nullptr, Alig,  nullptr, D_, D_);
  gemm16<0,0,false><<<dim3(P_/128, D_/64), 256, 0, stream>>>(prot16, D_, w1pT, D_, nullptr, Bprot, nullptr, D_, D_);
  imap_k<<<LP_/32, 256, 0, stream>>>(rbf16, w1rT, Alig, Bprot, ib1, iw2, ib2, outp + 3);
}
